// TransformerCrossBlock_41257455845974
// MI455X (gfx1250) — matrix operand order repaired and checked by reading, not compiled, not run on hardware
//
#include <hip/hip_runtime.h>

typedef __bf16 bf16_t;
typedef __attribute__((ext_vector_type(16))) __bf16 v16bf;
typedef __attribute__((ext_vector_type(8)))  __bf16 v8bf;
typedef __attribute__((ext_vector_type(8)))  float  v8f;

__device__ __forceinline__ bf16_t f2bf(float f) {
  unsigned u = __builtin_bit_cast(unsigned, f);
  unsigned r = (u + 0x7FFFu + ((u >> 16) & 1u)) >> 16;
  unsigned short s = (unsigned short)r;
  return __builtin_bit_cast(bf16_t, s);
}
__device__ __forceinline__ float bf2f(bf16_t b) {
  unsigned short s = __builtin_bit_cast(unsigned short, b);
  unsigned u = ((unsigned)s) << 16;
  return __builtin_bit_cast(float, u);
}

// CDNA5 async global->LDS copy (16B per lane), tracked by ASYNCcnt.
// lds_off is a byte offset into the wave's LDS allocation (dynamic-LDS base 0).
__device__ __forceinline__ void async_ld16(unsigned lds_off, const void* gsrc) {
  asm volatile("global_load_async_to_lds_b128 %0, %1, off"
               :: "v"(lds_off), "v"((unsigned long long)gsrc)
               : "memory");
}
// Wait until at most Nw async ops remain outstanding (in-order drain).
template <int Nw>
__device__ __forceinline__ void wait_async_le() {
  if constexpr (Nw == 0)
    asm volatile("s_wait_asynccnt 0" ::: "memory");
  else if constexpr (Nw == 2)
    asm volatile("s_wait_asynccnt 2" ::: "memory");
  else
    asm volatile("s_wait_asynccnt 4" ::: "memory");
}

// ---------------------------------------------------------------------------
// Elementwise fp32 -> bf16 conversion (weights / context), 4 elems per thread.
// ---------------------------------------------------------------------------
__global__ __launch_bounds__(256) void f2bf_kernel(
    const float* __restrict__ in, bf16_t* __restrict__ out, long long n) {
  long long i = ((long long)blockIdx.x * 256 + threadIdx.x) * 4;
  if (i + 3 < n) {
    float4 v = *(const float4*)(in + i);
    out[i + 0] = f2bf(v.x);
    out[i + 1] = f2bf(v.y);
    out[i + 2] = f2bf(v.z);
    out[i + 3] = f2bf(v.w);
  }
}

// ---------------------------------------------------------------------------
// LayerNorm over last dim (C), output bf16. One block per row.
// ---------------------------------------------------------------------------
__global__ __launch_bounds__(256) void ln_bf16_kernel(
    const float* __restrict__ x, const float* __restrict__ gamma,
    const float* __restrict__ beta, bf16_t* __restrict__ out, int C) {
  __shared__ float red[256];
  const int row = blockIdx.x;
  const int tid = threadIdx.x;
  const float* xr = x + (long long)row * C;
  float s = 0.f, sq = 0.f;
  for (int c = tid; c < C; c += 256) { float v = xr[c]; s += v; sq += v * v; }
  red[tid] = s; __syncthreads();
  for (int off = 128; off > 0; off >>= 1) {
    if (tid < off) red[tid] += red[tid + off];
    __syncthreads();
  }
  float mean = red[0] / (float)C;
  __syncthreads();
  red[tid] = sq; __syncthreads();
  for (int off = 128; off > 0; off >>= 1) {
    if (tid < off) red[tid] += red[tid + off];
    __syncthreads();
  }
  float var = red[0] / (float)C - mean * mean;
  float inv = rsqrtf(var + 1e-5f);
  bf16_t* orow = out + (long long)row * C;
  for (int c = tid; c < C; c += 256)
    orow[c] = f2bf((xr[c] - mean) * inv * gamma[c] + beta[c]);
}

// ---------------------------------------------------------------------------
// Tiled bf16 WMMA GEMM (all operands bf16), double-buffered + async-pipelined.
//   TRANSB=true : D[m,n] = sum_k A[m,k] * B[n,k]   (x @ W.T style)
//   TRANSB=false: D[m,n] = sum_k A[m,k] * B[k,n]   (attn @ V style)
// Block 128x128, BK=32, 8 waves (4x2), wave tile 32x64 (2x4 WMMA tiles).
// Tiles staged with GLOBAL_LOAD_ASYNC_TO_LDS_B128 into ping/pong LDS buffers;
// s_wait_asynccnt <= batch drains only the previous tile's loads, so the next
// tile's copy overlaps the current tile's WMMA work.
// Assumes M,N % 128 == 0, K % 32 == 0 (true for all uses here).
// ---------------------------------------------------------------------------
template <bool TRANSB, bool BIAS, bool RES, bool RELU_, bool OUTF_, bool OUTB_>
__global__ __launch_bounds__(256) void gemm_wmma_kernel(
    const bf16_t* __restrict__ A, const bf16_t* __restrict__ Bm,
    const float* __restrict__ bias, const float* __restrict__ residual,
    float* __restrict__ outF, bf16_t* __restrict__ outB,
    int M, int Nn, int K, int lda, int ldb, int ldc,
    long long strideA, long long strideB, long long strideC) {
  constexpr int LDSS = 40;            // 32 + 8 pad: bank spread, 16B aligned
  constexpr int TILE_E = 128 * LDSS;  // elems per (A or B) tile = 5120
  constexpr int BUF_E = 2 * TILE_E;   // elems per ping/pong buffer = 10240
  extern __shared__ __align__(16) char smem[];  // 2 * BUF_E * 2B = 40960 B
  bf16_t* sbase = (bf16_t*)smem;

  const int tid = threadIdx.x;
  const int lane = tid & 31;
  const int w = tid >> 5;   // 0..7
  const int wm = w >> 1;    // 0..3
  const int wn = w & 1;     // 0..1
  const int lm = lane & 15;
  const int hf = lane >> 4;

  const int bm = blockIdx.y * 128;
  const int bn = blockIdx.x * 128;
  const long long za = strideA * (long long)blockIdx.z;
  const long long zb = strideB * (long long)blockIdx.z;
  const long long zc = strideC * (long long)blockIdx.z;

  v8f zero = {};
  v8f acc[2][4];
  for (int i = 0; i < 2; ++i)
    for (int j = 0; j < 4; ++j) acc[i][j] = zero;

  // stage tile k0 into buffer buf (async for A, and for B when TRANSB)
  auto stage = [&](int k0, int buf) {
    const unsigned bufB = (unsigned)buf * (BUF_E * 2u);  // byte offset
    const int r0 = tid >> 2;            // 0..63
    const int c0 = (tid & 3) * 8;       // 0..24
#pragma unroll
    for (int g = 0; g < 2; ++g) {
      int r = r0 + g * 64;
      const bf16_t* src = A + za + (long long)(bm + r) * lda + k0 + c0;
      async_ld16(bufB + (unsigned)(r * LDSS + c0) * 2u, src);
    }
    if constexpr (TRANSB) {
#pragma unroll
      for (int g = 0; g < 2; ++g) {
        int r = r0 + g * 64;
        const bf16_t* src = Bm + zb + (long long)(bn + r) * ldb + k0 + c0;
        async_ld16(bufB + (unsigned)(TILE_E + r * LDSS + c0) * 2u, src);
      }
    } else {
      bf16_t* sB = sbase + buf * BUF_E + TILE_E;
      const int kk0 = tid >> 4;         // 0..15
      const int n0 = (tid & 15) * 8;    // 0..120
      for (int g = 0; g < 2; ++g) {
        int kk = kk0 + g * 16;
        const bf16_t* src = Bm + zb + (long long)(k0 + kk) * ldb + bn + n0;
#pragma unroll
        for (int e = 0; e < 8; ++e) sB[(n0 + e) * LDSS + kk] = src[e];
      }
    }
  };
  constexpr int BATCH = TRANSB ? 4 : 2;  // async ops per thread per stage

  stage(0, 0);
  const int KT = K / 32;
  for (int kt = 0; kt < KT; ++kt) {
    const int buf = kt & 1;
    if (kt + 1 < KT) {
      stage((kt + 1) * 32, buf ^ 1);   // prefetch next tile into other buffer
      wait_async_le<BATCH>();          // drain only the previous batch
    } else {
      wait_async_le<0>();
    }
    __syncthreads();

    const bf16_t* sA = sbase + buf * BUF_E;
    const bf16_t* sB = sA + TILE_E;
    v16bf af[2], bfr[4];
#pragma unroll
    for (int i = 0; i < 2; ++i) {
      // A 16x32 bf16 frag: lane m=lm, half hf; two contiguous 16B reads.
      int row = wm * 32 + i * 16 + lm;
      const bf16_t* p = &sA[row * LDSS + hf * 8];
      v8bf lo = *(const v8bf*)p;
      v8bf hi = *(const v8bf*)(p + 16);
#pragma unroll
      for (int e = 0; e < 8; ++e) { af[i][e] = lo[e]; af[i][e + 8] = hi[e]; }
    }
#pragma unroll
    for (int j = 0; j < 4; ++j) {
      // B 32x16 bf16 frag: lane n=lm, half hf; K = hf*8 + 0..7, 16 + hf*8 + 0..7.
      int nrow = wn * 64 + j * 16 + lm;
      const bf16_t* p = &sB[nrow * LDSS + hf * 8];
      v8bf lo = *(const v8bf*)p;
      v8bf hi = *(const v8bf*)(p + 16);
#pragma unroll
      for (int e = 0; e < 8; ++e) { bfr[j][e] = lo[e]; bfr[j][e + 8] = hi[e]; }
    }
#pragma unroll
    for (int i = 0; i < 2; ++i)
#pragma unroll
      for (int j = 0; j < 4; ++j)
        acc[i][j] = __builtin_amdgcn_wmma_f32_16x16x32_bf16(
            false, af[i], false, bfr[j], (short)0, acc[i][j], false, false);
    __syncthreads();
  }

  // ---- epilogue: D vgpr r -> (m = hf*8 + r, n = lm) within 16x16 tile ----
#pragma unroll
  for (int i = 0; i < 2; ++i) {
#pragma unroll
    for (int j = 0; j < 4; ++j) {
      const int n = bn + wn * 64 + j * 16 + lm;
      const int m0 = bm + wm * 32 + i * 16 + hf * 8;
      float bv = 0.f;
      if constexpr (BIAS) bv = bias[n];
      const long long base = zc + (long long)m0 * ldc + n;
#pragma unroll
      for (int r = 0; r < 8; ++r) {
        float v = acc[i][j][r] + bv;
        if constexpr (RELU_) v = fmaxf(v, 0.f);
        const long long idx = base + (long long)r * ldc;
        if constexpr (RES) v += residual[idx];
        if constexpr (OUTF_) outF[idx] = v;
        if constexpr (OUTB_) outB[idx] = f2bf(v);
      }
    }
  }
}

// ---------------------------------------------------------------------------
// Self-attention over the batch axis: per (n,h), 8x8 scores / softmax / out.
// One wave per (n,h); 8 waves per block. qkv is bf16 [B*N, 3C].
// ---------------------------------------------------------------------------
__global__ __launch_bounds__(256) void selfattn_kernel(
    const bf16_t* __restrict__ qkv, bf16_t* __restrict__ sa, int N, int C) {
  __shared__ float sq[8][8][64];
  __shared__ float sk[8][8][64];
  __shared__ float sv[8][8][64];
  __shared__ float ss[8][8][8];
  __shared__ float sp[8][8][8];
  const int tid = threadIdx.x;
  const int w = tid >> 5, lane = tid & 31;
  const int nh = blockIdx.x * 8 + w;
  const int n = nh >> 4, h = nh & 15;
  const int C3 = 3 * C;
  for (int idx = lane; idx < 512; idx += 32) {
    int i = idx >> 6, d = idx & 63;
    const bf16_t* p = qkv + (long long)(i * N + n) * C3 + h * 64 + d;
    sq[w][i][d] = bf2f(p[0]);
    sk[w][i][d] = bf2f(p[C]);
    sv[w][i][d] = bf2f(p[2 * C]);
  }
  __syncthreads();
  for (int t = 0; t < 2; ++t) {
    int pidx = lane + t * 32;
    int i = pidx >> 3, j = pidx & 7;
    float a = 0.f;
    for (int d = 0; d < 64; ++d) a += sq[w][i][d] * sk[w][j][d];
    ss[w][i][j] = a * 0.125f;  // SCALE = 64^-0.5
  }
  __syncthreads();
  for (int t = 0; t < 2; ++t) {
    int pidx = lane + t * 32;
    int i = pidx >> 3, j = pidx & 7;
    float mx = -3.4e38f;
    for (int jj = 0; jj < 8; ++jj) mx = fmaxf(mx, ss[w][i][jj]);
    float sum = 0.f;
    for (int jj = 0; jj < 8; ++jj) sum += __expf(ss[w][i][jj] - mx);
    sp[w][i][j] = __expf(ss[w][i][j] - mx) / sum;
  }
  __syncthreads();
  for (int idx = lane; idx < 512; idx += 32) {
    int i = idx >> 6, d = idx & 63;
    float o = 0.f;
    for (int j = 0; j < 8; ++j) o += sp[w][i][j] * sv[w][j][d];
    sa[(long long)(i * N + n) * C + h * 64 + d] = f2bf(o);
  }
}

// ---------------------------------------------------------------------------
// Row softmax on bf16, in place, with pre-scale. One block per row of length L.
// ---------------------------------------------------------------------------
__global__ __launch_bounds__(256) void softmax_bf16_kernel(
    bf16_t* __restrict__ s, int L, float scale) {
  __shared__ float red[256];
  bf16_t* p = s + (long long)blockIdx.x * L;
  const int tid = threadIdx.x;
  float mx = -3.4e38f;
  for (int c = tid; c < L; c += 256) mx = fmaxf(mx, bf2f(p[c]) * scale);
  red[tid] = mx; __syncthreads();
  for (int off = 128; off > 0; off >>= 1) {
    if (tid < off) red[tid] = fmaxf(red[tid], red[tid + off]);
    __syncthreads();
  }
  mx = red[0]; __syncthreads();
  float sum = 0.f;
  for (int c = tid; c < L; c += 256) sum += __expf(bf2f(p[c]) * scale - mx);
  red[tid] = sum; __syncthreads();
  for (int off = 128; off > 0; off >>= 1) {
    if (tid < off) red[tid] += red[tid + off];
    __syncthreads();
  }
  float inv = 1.f / red[0];
  for (int c = tid; c < L; c += 256)
    p[c] = f2bf(__expf(bf2f(p[c]) * scale - mx) * inv);
}

// ---------------------------------------------------------------------------
extern "C" void kernel_launch(void* const* d_in, const int* in_sizes, int n_in,
                              void* d_out, int out_size, void* d_ws, size_t ws_size,
                              hipStream_t stream) {
  (void)in_sizes; (void)n_in; (void)out_size; (void)ws_size;
  const float* x       = (const float*)d_in[0];
  const float* context = (const float*)d_in[1];
  const float* gamma   = (const float*)d_in[2];
  const float* beta    = (const float*)d_in[3];
  const float* w_in    = (const float*)d_in[4];
  const float* b_in    = (const float*)d_in[5];
  const float* w_out   = (const float*)d_in[6];
  const float* b_out   = (const float*)d_in[7];
  const float* wq      = (const float*)d_in[8];
  const float* bq      = (const float*)d_in[9];
  const float* wkv     = (const float*)d_in[10];
  const float* bkv     = (const float*)d_in[11];
  const float* w1      = (const float*)d_in[12];
  const float* b1      = (const float*)d_in[13];
  const float* w2      = (const float*)d_in[14];
  const float* b2      = (const float*)d_in[15];

  constexpr int B = 8, N = 2048, C = 1024, H = 16;
  constexpr int M = B * N;  // 16384 rows
  char* ws = (char*)d_ws;
  const size_t MB = 1024ull * 1024ull;
  // workspace layout (bump-allocated with reuse), ~482 MB total:
  bf16_t* xn   = (bf16_t*)(ws + 0);         // 32 MB : shared LN output
  bf16_t* big  = (bf16_t*)(ws + 32 * MB);   // 128 MB: qkv -> kv -> h1
  bf16_t* saq  = (bf16_t*)(ws + 160 * MB);  // 32 MB : sa -> q
  bf16_t* sc   = (bf16_t*)(ws + 192 * MB);  // 64 MB : scores/probs (in place)
  float*  x1   = (float*)(ws + 256 * MB);   // 64 MB
  float*  x2   = (float*)(ws + 320 * MB);   // 64 MB
  bf16_t* ctxb = (bf16_t*)(ws + 384 * MB);  // 64 MB : context bf16
  bf16_t* winb = (bf16_t*)(ws + 448 * MB);  // 6 MB
  bf16_t* woub = (bf16_t*)(ws + 454 * MB);  // 2 MB
  bf16_t* wqb  = (bf16_t*)(ws + 456 * MB);  // 2 MB
  bf16_t* wkvb = (bf16_t*)(ws + 458 * MB);  // 8 MB
  bf16_t* w1b  = (bf16_t*)(ws + 466 * MB);  // 8 MB
  bf16_t* w2b  = (bf16_t*)(ws + 474 * MB);  // 8 MB
  float*  out  = (float*)d_out;

  dim3 blk(256);
  constexpr unsigned GEMM_LDS = 2u * 2u * 128u * 40u * 2u;  // 40960 bytes

  // 0) one-time bf16 conversion of context + all weights
  auto cvt = [&](const float* src, bf16_t* dst, long long n) {
    f2bf_kernel<<<(unsigned)(n / 1024), blk, 0, stream>>>(src, dst, n);
  };
  cvt(context, ctxb, (long long)M * 2 * C);
  cvt(w_in, winb, (long long)3 * C * C);
  cvt(w_out, woub, (long long)C * C);
  cvt(wq, wqb, (long long)C * C);
  cvt(wkv, wkvb, (long long)2 * C * 2 * C);
  cvt(w1, w1b, (long long)4 * C * C);
  cvt(w2, w2b, (long long)C * 4 * C);

  // 1) xn = LN(x)
  ln_bf16_kernel<<<M, blk, 0, stream>>>(x, gamma, beta, xn, C);
  // 2) qkv = xn @ w_in.T + b_in  (bf16 out)
  gemm_wmma_kernel<true, true, false, false, false, true>
      <<<dim3(3 * C / 128, M / 128, 1), blk, GEMM_LDS, stream>>>(
      xn, winb, b_in, nullptr, nullptr, big,
      M, 3 * C, C, C, C, 3 * C, 0, 0, 0);
  // 3) self-attention over batch axis -> sa (bf16)
  selfattn_kernel<<<(N * H) / 8, blk, 0, stream>>>(big, saq, N, C);
  // 4) x1 = sa @ w_out.T + b_out + x
  gemm_wmma_kernel<true, true, true, false, true, false>
      <<<dim3(C / 128, M / 128, 1), blk, GEMM_LDS, stream>>>(
      saq, woub, b_out, x, x1, nullptr,
      M, C, C, C, C, C, 0, 0, 0);
  // 5) xn = LN(x1)
  ln_bf16_kernel<<<M, blk, 0, stream>>>(x1, gamma, beta, xn, C);
  // 6) q = xn @ wq.T + bq  (bf16 out, reuses sa buffer)
  gemm_wmma_kernel<true, true, false, false, false, true>
      <<<dim3(C / 128, M / 128, 1), blk, GEMM_LDS, stream>>>(
      xn, wqb, bq, nullptr, nullptr, saq,
      M, C, C, C, C, C, 0, 0, 0);
  // 7) kv = context @ wkv.T + bkv  (bf16 out)
  gemm_wmma_kernel<true, true, false, false, false, true>
      <<<dim3(2 * C / 128, M / 128, 1), blk, GEMM_LDS, stream>>>(
      ctxb, wkvb, bkv, nullptr, nullptr, big,
      M, 2 * C, 2 * C, 2 * C, 2 * C, 2 * C, 0, 0, 0);
  // 8) scores[b] = q[b] @ k[b].T  (head-summed contraction; batched)
  gemm_wmma_kernel<true, false, false, false, false, true>
      <<<dim3(N / 128, N / 128, B), blk, GEMM_LDS, stream>>>(
      saq, big, nullptr, nullptr, nullptr, sc,
      N, N, C, C, 2 * C, N,
      (long long)N * C, (long long)N * 2 * C, (long long)N * N);
  // 9) softmax rows (scale = D^-0.5 = 0.125), in place
  softmax_bf16_kernel<<<B * N, blk, 0, stream>>>(sc, N, 0.125f);
  // 10) x2[b] = attn[b] @ v[b] + x1[b]  (B non-transposed, batched)
  gemm_wmma_kernel<false, false, true, false, true, false>
      <<<dim3(C / 128, N / 128, B), blk, GEMM_LDS, stream>>>(
      sc, big + C, nullptr, x1, x2, nullptr,
      N, C, N, N, 2 * C, C,
      (long long)N * N, (long long)N * 2 * C, (long long)N * C);
  // 11) xn = LN(x2)
  ln_bf16_kernel<<<M, blk, 0, stream>>>(x2, gamma, beta, xn, C);
  // 12) h1 = relu(xn @ w1.T + b1)  (bf16 out)
  gemm_wmma_kernel<true, true, false, true, false, true>
      <<<dim3(4 * C / 128, M / 128, 1), blk, GEMM_LDS, stream>>>(
      xn, w1b, b1, nullptr, nullptr, big,
      M, 4 * C, C, C, C, 4 * C, 0, 0, 0);
  // 13) out = h1 @ w2.T + b2 + x2  (fp32, final)
  gemm_wmma_kernel<true, true, true, false, true, false>
      <<<dim3(C / 128, M / 128, 1), blk, GEMM_LDS, stream>>>(
      big, w2b, b2, x2, out, nullptr,
      M, C, 4 * C, 4 * C, 4 * C, C, 0, 0, 0);
}
